// DCRNNEncoder_60696477827517
// MI455X (gfx1250) — matrix operand order repaired and checked by reading, not compiled, not run on hardware
//
#include <hip/hip_runtime.h>
#include <stdint.h>

typedef __bf16 bf16;
typedef __attribute__((ext_vector_type(16))) __bf16 bf16x16;
typedef __attribute__((ext_vector_type(8)))  float  v8f;

#define T_STEPS 24
#define B_SZ    64
#define N_NODES 207
#define IN_DIM  2
#define H_DIM   64
#define NPK     256              // padded node count (mult of 64: M, N and K clean)
#define DP      128              // padded per-term feature dim (both layers)
#define K3      384              // dense K = 3*DP (mult of 64)
#define NH      (N_NODES*H_DIM)  // 13248
#define BNH     (B_SZ*NH)        // 847872

#define BM 64
#define BN 64
#define BK 64
#define LDS_PAD 8                // +8 bf16 per row: 144B stride -> ~2-way conflicts

union Frag { bf16x16 v; float4 q[2]; };
union LdU  { float4 q; bf16 e[8]; };

__device__ __forceinline__ float sigmoid_f(float x) {
  return 1.0f / (1.0f + __expf(-x));
}

// C[M,N] = A[M,K] * B[K,N]; A,B bf16 row-major, f32 WMMA accumulation.
// Caller guarantees: M,N multiples of tile coverage or store-guarded; K % 64 == 0;
// all K columns/rows valid (zero-padded) -> no load guards needed.
// mode 0: Cf(f32)=acc   mode 1: Cb(bf16)=acc   mode 2: Cb(bf16)=2*acc-E (Chebyshev)
__global__ __launch_bounds__(256)
void gemm_wmma_bf16(const bf16* __restrict__ A, int lda,
                    const bf16* __restrict__ Bg, int ldb,
                    int M, int N, int K,
                    float* __restrict__ Cf, bf16* __restrict__ Cb, int ldc,
                    const bf16* __restrict__ E, int ldE, int mode,
                    int64_t sA, int64_t sB, int64_t sC, int64_t sE)
{
  const int z = blockIdx.z;
  A  += (int64_t)z * sA;
  Bg += (int64_t)z * sB;
  if (Cf) Cf += (int64_t)z * sC;
  if (Cb) Cb += (int64_t)z * sC;
  if (E)  E  += (int64_t)z * sE;

  __shared__ __align__(16) bf16 As[2][BM][BK + LDS_PAD];   // [buf][m][k]
  __shared__ __align__(16) bf16 Bs[2][BN][BK + LDS_PAD];   // [buf][n][k] (transposed)

  const int tid  = threadIdx.x;
  const int wave = tid >> 5;
  const int lane = tid & 31;
  const int lr   = lane & 15;
  const int hi   = lane >> 4;

  const int m0 = blockIdx.y * BM;
  const int n0 = blockIdx.x * BN;
  const int wm = (wave & 3) * 16;    // 4 waves along M
  const int wn = (wave >> 2) * 32;   // 2 waves along N, 2 16-col tiles each

  v8f acc0 = {};
  v8f acc1 = {};

  // per-thread tile chunks: 2x float4 for A (rows am, am+32), 2x for B (k rows bk, bk+32)
  const int am  = tid >> 3;          // 0..31
  const int akc = (tid & 7) * 8;     // 16B k-chunk
  const int bk  = tid >> 3;
  const int bnc = (tid & 7) * 8;

  const bf16* Ar0 = A  + (int64_t)(m0 + am)      * lda + akc;
  const bf16* Ar1 = A  + (int64_t)(m0 + am + 32) * lda + akc;
  const bf16* Br0 = Bg + (int64_t)bk        * ldb + n0 + bnc;
  const bf16* Br1 = Bg + (int64_t)(bk + 32) * ldb + n0 + bnc;

  float4 ra0, ra1, rb0, rb1;

  auto stage = [&](int buf) {
    *reinterpret_cast<float4*>(&As[buf][am][akc])      = ra0;
    *reinterpret_cast<float4*>(&As[buf][am + 32][akc]) = ra1;
    LdU t0, t1; t0.q = rb0; t1.q = rb1;
    #pragma unroll
    for (int j = 0; j < 8; ++j) {
      Bs[buf][bnc + j][bk]      = t0.e[j];
      Bs[buf][bnc + j][bk + 32] = t1.e[j];
    }
  };

  auto compute = [&](int buf) {
    #pragma unroll
    for (int ks = 0; ks < 2; ++ks) {
      const int kb = ks * 32;
      Frag af, bf0, bf1;
      // A frag (16-bit A 16x32): lane half hi -> K 0..15 / 16..31 (contiguous, matches B)
      af.q[0]  = *reinterpret_cast<const float4*>(&As[buf][wm + lr][kb + hi * 16]);
      af.q[1]  = *reinterpret_cast<const float4*>(&As[buf][wm + lr][kb + hi * 16 + 8]);
      // B frag (lanes = N column, K 0..15 / 16..31 by lane half)
      bf0.q[0] = *reinterpret_cast<const float4*>(&Bs[buf][wn + lr][kb + hi * 16]);
      bf0.q[1] = *reinterpret_cast<const float4*>(&Bs[buf][wn + lr][kb + hi * 16 + 8]);
      bf1.q[0] = *reinterpret_cast<const float4*>(&Bs[buf][wn + 16 + lr][kb + hi * 16]);
      bf1.q[1] = *reinterpret_cast<const float4*>(&Bs[buf][wn + 16 + lr][kb + hi * 16 + 8]);
      acc0 = __builtin_amdgcn_wmma_f32_16x16x32_bf16(false, af.v, false, bf0.v,
                                                     (short)0, acc0, false, false);
      acc1 = __builtin_amdgcn_wmma_f32_16x16x32_bf16(false, af.v, false, bf1.v,
                                                     (short)0, acc1, false, false);
    }
  };

  const int nt = K >> 6;   // K / 64

  // prologue: tile 0 -> buffer 0
  ra0 = *reinterpret_cast<const float4*>(Ar0);
  ra1 = *reinterpret_cast<const float4*>(Ar1);
  rb0 = *reinterpret_cast<const float4*>(Br0);
  rb1 = *reinterpret_cast<const float4*>(Br1);
  stage(0);
  __syncthreads();

  for (int it = 0; it < nt; ++it) {
    const int cur = it & 1;
    if (it + 1 < nt) {
      Ar0 += BK; Ar1 += BK;
      Br0 += (int64_t)BK * ldb; Br1 += (int64_t)BK * ldb;
      ra0 = *reinterpret_cast<const float4*>(Ar0);
      ra1 = *reinterpret_cast<const float4*>(Ar1);
      rb0 = *reinterpret_cast<const float4*>(Br0);
      rb1 = *reinterpret_cast<const float4*>(Br1);
      if (it + 2 < nt) {
        __builtin_prefetch(Ar0 + BK, 0, 1);
        __builtin_prefetch(Br0 + (int64_t)BK * ldb, 0, 1);
      }
    }
    compute(cur);
    if (it + 1 < nt) {
      stage(cur ^ 1);
      __syncthreads();
    }
  }

  // C layout: VGPR v -> row wm + hi*8 + v, col = lane%16 (+16 for second tile)
  #pragma unroll
  for (int v = 0; v < 8; ++v) {
    int gm = m0 + wm + hi * 8 + v;
    if (gm >= M) continue;
    int gn0 = n0 + wn + lr;
    int gn1 = gn0 + 16;
    float a0 = acc0[v], a1 = acc1[v];
    if (mode == 0) {
      if (gn0 < N) Cf[(int64_t)gm * ldc + gn0] = a0;
      if (gn1 < N) Cf[(int64_t)gm * ldc + gn1] = a1;
    } else if (mode == 1) {
      if (gn0 < N) Cb[(int64_t)gm * ldc + gn0] = (bf16)a0;
      if (gn1 < N) Cb[(int64_t)gm * ldc + gn1] = (bf16)a1;
    } else {
      if (gn0 < N) Cb[(int64_t)gm * ldc + gn0] =
          (bf16)(2.0f * a0 - (float)E[(int64_t)gm * ldE + gn0]);
      if (gn1 < N) Cb[(int64_t)gm * ldc + gn1] =
          (bf16)(2.0f * a1 - (float)E[(int64_t)gm * ldE + gn1]);
    }
  }
}

// ---- prep: pad S (207x207 -> 256x256 bf16, zero slack) ----
__global__ void pad_S_kernel(const float* __restrict__ S, bf16* __restrict__ Sb) {
  int idx = blockIdx.x * 256 + threadIdx.x;
  if (idx >= NPK * NPK) return;
  int r = idx / NPK, c = idx - r * NPK;
  float v = (r < N_NODES && c < N_NODES) ? S[r * N_NODES + c] : 0.0f;
  Sb[idx] = (bf16)v;
}

// ---- prep: pad weights [(3*blk) x O] -> [(3*DP) x O] bf16 (per-term row pad) ----
__global__ void pad_W_kernel(const float* __restrict__ W, int blk, int O,
                             bf16* __restrict__ Wp) {
  int idx = blockIdx.x * 256 + threadIdx.x;
  if (idx >= 3 * DP * O) return;
  int row = idx / O, col = idx - row * O;
  int term = row / DP, d = row - term * DP;
  float v = (d < blk) ? W[(term * blk + d) * O + col] : 0.0f;
  Wp[idx] = (bf16)v;
}

// ---- build X0 = concat(x_t, h) into term-0 slot of padded concat buffer ----
__global__ void pack_x0_kernel(const float* __restrict__ x, int64_t sxb, int sxn, int in_dim,
                               const float* __restrict__ h,
                               bf16* __restrict__ X) {
  int b = blockIdx.y;
  int idx = blockIdx.x * 256 + threadIdx.x;
  if (idx >= NPK * DP) return;
  int n = idx / DP, d = idx - n * DP;
  float v = 0.0f;
  if (n < N_NODES) {
    if (d < in_dim)              v = x[(int64_t)b * sxb + (int64_t)n * sxn + d];
    else if (d < in_dim + H_DIM) v = h[(int64_t)b * NH + n * H_DIM + (d - in_dim)];
  }
  X[((int64_t)b * NPK + n) * K3 + d] = (bf16)v;
}

// ---- gates: r,u = sigmoid(G+bg); save u; build Xc0 = concat(x, r*h) ----
__global__ void gates_act_kernel(const float* __restrict__ G, const float* __restrict__ bg,
                                 const float* __restrict__ x, int64_t sxb, int sxn, int in_dim,
                                 const float* __restrict__ h,
                                 float* __restrict__ U,
                                 bf16* __restrict__ Xc) {
  int b = blockIdx.y;
  int idx = blockIdx.x * 256 + threadIdx.x;
  if (idx >= NPK * DP) return;
  int n = idx / DP, d = idx - n * DP;
  float v = 0.0f;
  if (n < N_NODES) {
    if (d < in_dim) {
      v = x[(int64_t)b * sxb + (int64_t)n * sxn + d];
    } else if (d < in_dim + H_DIM) {
      int j = d - in_dim;
      int64_t grow = ((int64_t)b * NPK + n) * 128;
      float r = sigmoid_f(G[grow + j] + bg[j]);
      float u = sigmoid_f(G[grow + 64 + j] + bg[64 + j]);
      U[((int64_t)b * N_NODES + n) * H_DIM + j] = u;
      v = r * h[(int64_t)b * NH + n * H_DIM + j];
    }
  }
  Xc[((int64_t)b * NPK + n) * K3 + d] = (bf16)v;
}

// ---- h = u*h + (1-u)*tanh(Cc + bc) ----
__global__ void cand_update_kernel(const float* __restrict__ Cc, const float* __restrict__ bc,
                                   const float* __restrict__ U,
                                   float* __restrict__ h) {
  int b = blockIdx.y;
  int idx = blockIdx.x * 256 + threadIdx.x;
  if (idx >= NH) return;
  int n = idx / H_DIM, j = idx - n * H_DIM;
  float c = tanhf(Cc[((int64_t)b * NPK + n) * H_DIM + j] + bc[j]);
  float u = U[(int64_t)b * NH + idx];
  int64_t hidx = (int64_t)b * NH + idx;
  float hv = h[hidx];
  h[hidx] = u * hv + (1.0f - u) * c;
}

extern "C" void kernel_launch(void* const* d_in, const int* in_sizes, int n_in,
                              void* d_out, int out_size, void* d_ws, size_t ws_size,
                              hipStream_t stream) {
  (void)in_sizes; (void)n_in; (void)out_size; (void)ws_size;
  const float* inputs  = (const float*)d_in[0];
  const float* h_init  = (const float*)d_in[1];
  const float* S       = (const float*)d_in[2];
  const float* w_gate0 = (const float*)d_in[3];
  const float* b_gate0 = (const float*)d_in[4];
  const float* w_cand0 = (const float*)d_in[5];
  const float* b_cand0 = (const float*)d_in[6];
  const float* w_gate1 = (const float*)d_in[7];
  const float* b_gate1 = (const float*)d_in[8];
  const float* w_cand1 = (const float*)d_in[9];
  const float* b_cand1 = (const float*)d_in[10];
  float* out = (float*)d_out;

  char* ws = (char*)d_ws;
  size_t off = 0;
  auto alloc = [&](size_t bytes) -> char* {
    char* p = ws + off;
    off = (off + bytes + 255) & ~(size_t)255;
    return p;
  };

  bf16*  Sb  = (bf16*) alloc((size_t)NPK * NPK * 2);        // 256x256 zero-padded
  bf16*  Wg0 = (bf16*) alloc((size_t)K3 * 128 * 2);
  bf16*  Wc0 = (bf16*) alloc((size_t)K3 * 64  * 2);
  bf16*  Wg1 = (bf16*) alloc((size_t)K3 * 128 * 2);
  bf16*  Wc1 = (bf16*) alloc((size_t)K3 * 64  * 2);
  bf16*  Xg  = (bf16*) alloc((size_t)B_SZ * NPK * K3 * 2);  // [B,256,384]
  bf16*  Xc  = (bf16*) alloc((size_t)B_SZ * NPK * K3 * 2);
  float* G   = (float*)alloc((size_t)B_SZ * NPK * 128 * 4);
  float* Cc  = (float*)alloc((size_t)B_SZ * NPK * 64  * 4);
  float* U   = (float*)alloc((size_t)BNH * 4);
  float* h0  = (float*)alloc((size_t)BNH * 4);
  float* h1  = (float*)alloc((size_t)BNH * 4);

  // one-time prep (re-run every call: deterministic)
  pad_S_kernel<<<(NPK*NPK + 255)/256, 256, 0, stream>>>(S, Sb);
  pad_W_kernel<<<(3*DP*128 + 255)/256, 256, 0, stream>>>(w_gate0,  66, 128, Wg0);
  pad_W_kernel<<<(3*DP*64  + 255)/256, 256, 0, stream>>>(w_cand0,  66,  64, Wc0);
  pad_W_kernel<<<(3*DP*128 + 255)/256, 256, 0, stream>>>(w_gate1, 128, 128, Wg1);
  pad_W_kernel<<<(3*DP*64  + 255)/256, 256, 0, stream>>>(w_cand1, 128,  64, Wc1);
  hipMemcpyAsync(h0, h_init,       (size_t)BNH * 4, hipMemcpyDeviceToDevice, stream);
  hipMemcpyAsync(h1, h_init + BNH, (size_t)BNH * 4, hipMemcpyDeviceToDevice, stream);

  const int     Mdense = B_SZ * NPK;            // 16384 rows (mult of 64)
  const int64_t sX     = (int64_t)NPK * K3;     // per-batch stride (elements)

  for (int t = 0; t < T_STEPS; ++t) {
    for (int layer = 0; layer < 2; ++layer) {
      const float* x; int64_t sxb; int sxn, in_dim;
      const bf16 *Wg, *Wc; const float *bg, *bc; float* h;
      if (layer == 0) {
        x = inputs + (int64_t)t * B_SZ * N_NODES * IN_DIM;
        sxb = (int64_t)N_NODES * IN_DIM; sxn = IN_DIM; in_dim = IN_DIM;
        Wg = Wg0; Wc = Wc0; bg = b_gate0; bc = b_cand0; h = h0;
      } else {
        x = h0; sxb = NH; sxn = H_DIM; in_dim = H_DIM;
        Wg = Wg1; Wc = Wc1; bg = b_gate1; bc = b_cand1; h = h1;
      }

      dim3 gPack((NPK * DP + 255)/256, B_SZ);
      pack_x0_kernel<<<gPack, 256, 0, stream>>>(x, sxb, sxn, in_dim, h, Xg);

      dim3 gDiff(DP/BN, NPK/BM, B_SZ);   // (2, 4, 64)
      // X1 = S @ X0  (batched over B via blockIdx.z; S shared, strideA=0)
      gemm_wmma_bf16<<<gDiff, 256, 0, stream>>>(Sb, NPK, Xg, K3, NPK, DP, NPK,
                                                nullptr, Xg + DP, K3,
                                                nullptr, 0, 1, 0, sX, sX, 0);
      // X2 = 2*S@X1 - X0 (fused Chebyshev epilogue)
      gemm_wmma_bf16<<<gDiff, 256, 0, stream>>>(Sb, NPK, Xg + DP, K3, NPK, DP, NPK,
                                                nullptr, Xg + 2*DP, K3,
                                                Xg, K3, 2, 0, sX, sX, sX);
      // G = [X0|X1|X2] @ Wg
      dim3 gDg(128/BN, Mdense/BM, 1);
      gemm_wmma_bf16<<<gDg, 256, 0, stream>>>(Xg, K3, Wg, 128, Mdense, 128, K3,
                                              G, nullptr, 128,
                                              nullptr, 0, 0, 0, 0, 0, 0);
      // sigmoid gates, keep u, build candidate concat term-0 with r*h
      gates_act_kernel<<<gPack, 256, 0, stream>>>(G, bg, x, sxb, sxn, in_dim, h, U, Xc);
      // candidate diffusion
      gemm_wmma_bf16<<<gDiff, 256, 0, stream>>>(Sb, NPK, Xc, K3, NPK, DP, NPK,
                                                nullptr, Xc + DP, K3,
                                                nullptr, 0, 1, 0, sX, sX, 0);
      gemm_wmma_bf16<<<gDiff, 256, 0, stream>>>(Sb, NPK, Xc + DP, K3, NPK, DP, NPK,
                                                nullptr, Xc + 2*DP, K3,
                                                Xc, K3, 2, 0, sX, sX, sX);
      // Cc = [Xc0|Xc1|Xc2] @ Wc
      dim3 gDc(64/BN, Mdense/BM, 1);
      gemm_wmma_bf16<<<gDc, 256, 0, stream>>>(Xc, K3, Wc, 64, Mdense, 64, K3,
                                              Cc, nullptr, 64,
                                              nullptr, 0, 0, 0, 0, 0, 0);
      dim3 gUpd((NH + 255)/256, B_SZ);
      cand_update_kernel<<<gUpd, 256, 0, stream>>>(Cc, bc, U, h);
    }
    // cur[t] = h1
    hipMemcpyAsync(out + (int64_t)(2 + t) * BNH, h1, (size_t)BNH * 4,
                   hipMemcpyDeviceToDevice, stream);
  }
  // output_hidden = stack(h0, h1)
  hipMemcpyAsync(out,       h0, (size_t)BNH * 4, hipMemcpyDeviceToDevice, stream);
  hipMemcpyAsync(out + BNH, h1, (size_t)BNH * 4, hipMemcpyDeviceToDevice, stream);
}
